// Mspe_33621003993817
// MI455X (gfx1250) — compile-verified, matrix operand order checked by reading, not run on hardware
//
#include <hip/hip_runtime.h>
#include <hip/hip_bf16.h>
#include <math.h>

// ---------------------------------------------------------------------------
// MST-style mask-guided spectral attention for MI455X (gfx1250, wave32).
// Heavy GEMMs use V_WMMA_F32_16X16X32_BF16 (f32 accumulate).
// ---------------------------------------------------------------------------

#define BB   4
#define HH   128
#define WW   128
#define CC   256
#define NN   (HH*WW)          // 16384 tokens per image
#define BN   (BB*NN)          // 65536 total rows
#define HEADS 8
#define DH    32

typedef __attribute__((ext_vector_type(16))) __bf16 v16bf;
typedef __attribute__((ext_vector_type(8)))  float  v8f;
typedef __attribute__((ext_vector_type(4)))  unsigned int v4u;

union Frag { v16bf v; v4u u[2]; };

// ---------------------------------------------------------------------------
// WMMA GEMM:  Out[M,256] = A[M,256](bf16) * W[256,256] (+bias)
// Wt is W pre-transposed: Wt[n*256 + k] = W[k,n]  (bf16, row-major [N,K]).
// Block: 256 threads = 8 waves. Block tile: 64 rows x 256 cols.
//   wave wy (0..3) -> 16-row subtile, wave wx (0..1) -> 128-col half,
//   each wave: 8 accumulators (16x16) x 8 K-steps of 32.
// A/B fragment (16-bit, 16x32): lane m = lane&15, half = lane>>4,
//   elems 0..7  = K + half*8 + (0..7)
//   elems 8..15 = K + 16 + half*8 + (0..7)
// C/D: lane -> n = lane&15, VGPR j -> m = (lane>>4)*8 + j.
// ---------------------------------------------------------------------------
__global__ __launch_bounds__(256)
void gemm_bf16_wmma(const __bf16* __restrict__ A,
                    const __bf16* __restrict__ Wt,
                    const float*  __restrict__ bias,
                    float* __restrict__ Out)
{
    const int K = 256, Nc = 256;
    const int lane = threadIdx.x & 31;
    const int wid  = threadIdx.x >> 5;
    const int wy   = wid & 3;
    const int wx   = wid >> 2;
    const int row0 = blockIdx.x * 64 + wy * 16;
    const int col0 = wx * 128;
    const int m    = lane & 15;
    const int half = lane >> 4;

    v8f acc[8] = {};

    const __bf16* aRow = A + (size_t)(row0 + m) * K + half * 8;

    for (int kb = 0; kb < K; kb += 32) {
        Frag af;
        af.u[0] = *(const v4u*)(aRow + kb);
        af.u[1] = *(const v4u*)(aRow + kb + 16);
#pragma unroll
        for (int t = 0; t < 8; ++t) {
            const int n = col0 + t * 16 + m;
            const __bf16* bRow = Wt + (size_t)n * K + half * 8 + kb;
            Frag bf;
            bf.u[0] = *(const v4u*)(bRow);
            bf.u[1] = *(const v4u*)(bRow + 16);
            acc[t] = __builtin_amdgcn_wmma_f32_16x16x32_bf16(
                        false, af.v, false, bf.v, (short)0, acc[t], false, false);
        }
    }

#pragma unroll
    for (int t = 0; t < 8; ++t) {
        const int n = col0 + t * 16 + (lane & 15);
        const float bv = bias ? bias[n] : 0.0f;
#pragma unroll
        for (int j = 0; j < 8; ++j) {
            const int mr = row0 + half * 8 + j;
            Out[(size_t)mr * Nc + n] = acc[t][j] + bv;
        }
    }
}

// --------------------------- small helper kernels ---------------------------

__global__ void k_zero(float* p, int count) {
    int i = blockIdx.x * blockDim.x + threadIdx.x;
    if (i < count) p[i] = 0.0f;
}

__global__ void k_cast_bf16(const float* __restrict__ in, __bf16* __restrict__ out, int count) {
    int i = blockIdx.x * blockDim.x + threadIdx.x;
    if (i < count) out[i] = (__bf16)in[i];
}

// Wt[n*256+k] = W[k*256+n]  (for x@W style GEMMs)
__global__ void k_transpose_w(const float* __restrict__ W, __bf16* __restrict__ Wt) {
    int k = threadIdx.x;        // 0..255
    int n = blockIdx.x;         // 0..255
    Wt[n * 256 + k] = (__bf16)W[k * 256 + n];
}

// maskT[p*256+c] = mask[c*16384+p]   (mask is [1,C,H,W])
__global__ void k_mask_transpose(const float* __restrict__ mask, __bf16* __restrict__ maskT) {
    int i = blockIdx.x * blockDim.x + threadIdx.x;
    if (i >= NN * CC) return;
    int c = i & 255;
    int p = i >> 8;
    maskT[i] = (__bf16)mask[c * NN + p];
}

// depthwise 5x5 (pad 2) on m2[HW,C] -> sigmoid -> maskemb = m1*(1+sig)
__global__ void k_dw5_sigmoid_emb(const float* __restrict__ m1,
                                  const float* __restrict__ m2,
                                  const float* __restrict__ dw,   // [C,25]
                                  const float* __restrict__ db,   // [C]
                                  float* __restrict__ emb)
{
    int i = blockIdx.x * blockDim.x + threadIdx.x;
    if (i >= NN * CC) return;
    int c = i & 255;
    int p = i >> 8;
    int h = p >> 7, w = p & 127;
    float acc = db[c];
#pragma unroll
    for (int dy = 0; dy < 5; ++dy) {
        int hh = h + dy - 2;
        if (hh < 0 || hh >= HH) continue;
#pragma unroll
        for (int dx = 0; dx < 5; ++dx) {
            int ww = w + dx - 2;
            if (ww < 0 || ww >= WW) continue;
            acc += m2[((hh << 7) + ww) * 256 + c] * dw[c * 25 + dy * 5 + dx];
        }
    }
    float s = 1.0f / (1.0f + expf(-acc));
    emb[i] = m1[i] * (1.0f + s);
}

// column sum-of-squares for q and k: acc[b*256+c] += sum_n x[b,n,c]^2
__global__ void k_colnorm_partial(const float* __restrict__ q,
                                  const float* __restrict__ k,
                                  float* __restrict__ accQ,
                                  float* __restrict__ accK)
{
    int b   = blockIdx.x >> 6;          // 0..3
    int seg = blockIdx.x & 63;          // 0..63, 256 rows each
    int c   = threadIdx.x;              // 0..255
    float sq = 0.f, sk = 0.f;
    size_t base = ((size_t)b * NN + (size_t)seg * 256) * 256 + c;
    for (int r = 0; r < 256; ++r) {
        float a = q[base + (size_t)r * 256];
        float d = k[base + (size_t)r * 256];
        sq += a * a;
        sk += d * d;
    }
    atomicAdd(&accQ[b * 256 + c], sq);
    atomicAdd(&accK[b * 256 + c], sk);
}

// Gram: G[b,h,i,j] += sum_n k[b,n,h*32+i] * q[b,n,h*32+j]
__global__ __launch_bounds__(256)
void k_gram(const float* __restrict__ q, const float* __restrict__ k,
            float* __restrict__ G)
{
    __shared__ float kT[64][32];
    __shared__ float qT[64][32];
    int bh  = blockIdx.x >> 3;          // 0..31 -> (b,h)
    int seg = blockIdx.x & 7;           // 0..7, 2048 rows each
    int b = bh >> 3, h = bh & 7;
    int t = threadIdx.x;
    int j = t & 31;
    int i0 = t >> 5;                    // 0..7
    float acc[4] = {0.f, 0.f, 0.f, 0.f};

    for (int chunk = 0; chunk < 32; ++chunk) {
        int n0 = seg * 2048 + chunk * 64;
        // cooperative load of 64x32 tiles (8 elems/thread each)
        for (int e = t; e < 64 * 32; e += 256) {
            int r = e >> 5, col = e & 31;
            size_t src = ((size_t)b * NN + n0 + r) * 256 + h * 32 + col;
            kT[r][col] = k[src];
            qT[r][col] = q[src];
        }
        __syncthreads();
        for (int r = 0; r < 64; ++r) {
            float qv = qT[r][j];
#pragma unroll
            for (int a = 0; a < 4; ++a)
                acc[a] += kT[r][i0 + 8 * a] * qv;
        }
        __syncthreads();
    }
#pragma unroll
    for (int a = 0; a < 4; ++a)
        atomicAdd(&G[((bh * 32) + i0 + 8 * a) * 32 + j], acc[a]);
}

// softmax over j of  G[i,j]/(|k_i||q_j|) * rescale[h]
__global__ void k_softmax(const float* __restrict__ G,
                          const float* __restrict__ accQ,
                          const float* __restrict__ accK,
                          const float* __restrict__ rescale,
                          float* __restrict__ attn)
{
    int t = blockIdx.x * blockDim.x + threadIdx.x;
    if (t >= BB * HEADS * DH) return;
    int b = t >> 8, h = (t >> 5) & 7, i = t & 31;
    float nk = fmaxf(sqrtf(accK[b * 256 + h * 32 + i]), 1e-12f);
    float rs = rescale[h];
    const float* g = G + (size_t)t * 32;
    float row[32], mx = -1e30f;
#pragma unroll
    for (int j = 0; j < 32; ++j) {
        float nq = fmaxf(sqrtf(accQ[b * 256 + h * 32 + j]), 1e-12f);
        row[j] = g[j] / (nk * nq) * rs;
        mx = fmaxf(mx, row[j]);
    }
    float sum = 0.f;
#pragma unroll
    for (int j = 0; j < 32; ++j) { row[j] = expf(row[j] - mx); sum += row[j]; }
    float inv = 1.0f / sum;
    float* o = attn + (size_t)t * 32;
#pragma unroll
    for (int j = 0; j < 32; ++j) o[j] = row[j] * inv;
}

// v_s[b,n,c] = v[b,n,c] * maskemb[n,c]
__global__ void k_vscale(const float* __restrict__ v, const float* __restrict__ emb,
                         float* __restrict__ vs)
{
    int i = blockIdx.x * blockDim.x + threadIdx.x;
    if (i >= BN * CC) return;
    int c = i & 255;
    int n = (i >> 8) & (NN - 1);
    vs[i] = v[i] * emb[n * 256 + c];
}

// out_t[b,n,h*32+i] = sum_j attn[b,h,i,j] * vs[b,n,h*32+j]
__global__ void k_headmix(const float* __restrict__ attn, const float* __restrict__ vs,
                          float* __restrict__ outT)
{
    int idx = blockIdx.x * blockDim.x + threadIdx.x;
    if (idx >= BN * CC) return;
    int c = idx & 255;
    int n = (idx >> 8) & (NN - 1);
    int b = idx >> 22;
    int h = c >> 5, i = c & 31;
    const float* a  = attn + (((size_t)(b * 8 + h) * 32) + i) * 32;
    const float* vv = vs + (((size_t)b * NN + n) * 256) + h * 32;
    float acc = 0.f;
#pragma unroll
    for (int j = 0; j < 32; ++j) acc += a[j] * vv[j];
    outT[idx] = acc;
}

__device__ __forceinline__ float gelu_exact(float x) {
    return 0.5f * x * (1.0f + erff(x * 0.70710678118654752f));
}

// depthwise 3x3 (pad 1) + exact GELU on v_inp [b,h,w,c]
__global__ void k_band1(const float* __restrict__ v, const float* __restrict__ w1, // [C,9]
                        float* __restrict__ y1)
{
    int idx = blockIdx.x * blockDim.x + threadIdx.x;
    if (idx >= BN * CC) return;
    int c = idx & 255;
    int p = (idx >> 8) & (NN - 1);
    int b = idx >> 22;
    int h = p >> 7, w = p & 127;
    float acc = 0.f;
#pragma unroll
    for (int dy = 0; dy < 3; ++dy) {
        int hh = h + dy - 1;
        if (hh < 0 || hh >= HH) continue;
#pragma unroll
        for (int dx = 0; dx < 3; ++dx) {
            int ww = w + dx - 1;
            if (ww < 0 || ww >= WW) continue;
            acc += v[(((size_t)b * NN + (hh << 7) + ww) << 8) + c] * w1[c * 9 + dy * 3 + dx];
        }
    }
    y1[idx] = gelu_exact(acc);
}

// depthwise 3x3 (pad 1), accumulate into d_out
__global__ void k_band2(const float* __restrict__ y1, const float* __restrict__ w2, // [C,9]
                        float* __restrict__ out)
{
    int idx = blockIdx.x * blockDim.x + threadIdx.x;
    if (idx >= BN * CC) return;
    int c = idx & 255;
    int p = (idx >> 8) & (NN - 1);
    int b = idx >> 22;
    int h = p >> 7, w = p & 127;
    float acc = 0.f;
#pragma unroll
    for (int dy = 0; dy < 3; ++dy) {
        int hh = h + dy - 1;
        if (hh < 0 || hh >= HH) continue;
#pragma unroll
        for (int dx = 0; dx < 3; ++dx) {
            int ww = w + dx - 1;
            if (ww < 0 || ww >= WW) continue;
            acc += y1[(((size_t)b * NN + (hh << 7) + ww) << 8) + c] * w2[c * 9 + dy * 3 + dx];
        }
    }
    out[idx] += acc;
}

// ---------------------------------------------------------------------------

extern "C" void kernel_launch(void* const* d_in, const int* in_sizes, int n_in,
                              void* d_out, int out_size, void* d_ws, size_t ws_size,
                              hipStream_t stream)
{
    const float* x       = (const float*)d_in[0];   // [B,H,W,C] == [BN,C]
    const float* mask    = (const float*)d_in[1];   // [1,C,H,W]
    const float* wq      = (const float*)d_in[2];
    const float* wk      = (const float*)d_in[3];
    const float* wv      = (const float*)d_in[4];
    const float* rescale = (const float*)d_in[5];
    const float* wproj   = (const float*)d_in[6];
    const float* bproj   = (const float*)d_in[7];
    const float* mg_w1   = (const float*)d_in[8];
    const float* mg_b1   = (const float*)d_in[9];
    const float* mg_w2   = (const float*)d_in[10];
    const float* mg_b2   = (const float*)d_in[11];
    const float* mg_dw   = (const float*)d_in[12];
    const float* mg_db   = (const float*)d_in[13];
    const float* be_w1   = (const float*)d_in[14];
    const float* be_w2   = (const float*)d_in[15];
    float* out = (float*)d_out;

    // ---- workspace carve-out ----
    char* p = (char*)d_ws;
    auto alloc = [&](size_t bytes) -> void* {
        void* r = (void*)p;
        p += (bytes + 255) & ~(size_t)255;
        return r;
    };
    __bf16* x_bf   = (__bf16*)alloc((size_t)BN * CC * 2);   // 32 MB (later reused: out_t bf16)
    __bf16* wq_t   = (__bf16*)alloc(65536 * 2);
    __bf16* wk_t   = (__bf16*)alloc(65536 * 2);
    __bf16* wv_t   = (__bf16*)alloc(65536 * 2);
    __bf16* wp_t   = (__bf16*)alloc(65536 * 2);
    __bf16* w1_bf  = (__bf16*)alloc(65536 * 2);
    __bf16* w2_bf  = (__bf16*)alloc(65536 * 2);
    float*  q      = (float*)alloc((size_t)BN * CC * 4);    // 64 MB (reused: v_s, then y1)
    float*  k      = (float*)alloc((size_t)BN * CC * 4);    // 64 MB (reused: out_t)
    float*  v      = (float*)alloc((size_t)BN * CC * 4);    // 64 MB
    __bf16* maskT  = (__bf16*)alloc((size_t)NN * CC * 2);   // 8 MB
    float*  m1     = (float*)alloc((size_t)NN * CC * 4);    // 16 MB
    __bf16* m1_bf  = (__bf16*)alloc((size_t)NN * CC * 2);   // 8 MB
    float*  m2     = (float*)alloc((size_t)NN * CC * 4);    // 16 MB
    float*  emb    = (float*)alloc((size_t)NN * CC * 4);    // 16 MB
    float*  accQ   = (float*)alloc(BB * CC * 4);
    float*  accK   = (float*)alloc(BB * CC * 4);
    float*  G      = (float*)alloc(BB * HEADS * DH * DH * 4);
    float*  attn   = (float*)alloc(BB * HEADS * DH * DH * 4);
    (void)ws_size; (void)n_in; (void)in_sizes; (void)out_size;

    const int ELN = BN * CC;                 // 16,777,216
    dim3 blk(256);
    dim3 grdFull(ELN / 256);                 // 65536 blocks
    dim3 grdMask((NN * CC) / 256);           // 16384 blocks

    // 0. zero the atomic accumulators (accQ, accK, G are contiguous)
    k_zero<<<dim3((BB * CC * 2 + BB * HEADS * DH * DH + 255) / 256), blk, 0, stream>>>(
        accQ, BB * CC * 2 + BB * HEADS * DH * DH);

    // 1. conversions
    k_cast_bf16<<<grdFull, blk, 0, stream>>>(x, x_bf, ELN);
    k_transpose_w<<<dim3(256), blk, 0, stream>>>(wq, wq_t);
    k_transpose_w<<<dim3(256), blk, 0, stream>>>(wk, wk_t);
    k_transpose_w<<<dim3(256), blk, 0, stream>>>(wv, wv_t);
    k_transpose_w<<<dim3(256), blk, 0, stream>>>(wproj, wp_t);
    k_cast_bf16<<<dim3(65536 / 256), blk, 0, stream>>>(mg_w1, w1_bf, 65536);
    k_cast_bf16<<<dim3(65536 / 256), blk, 0, stream>>>(mg_w2, w2_bf, 65536);
    k_mask_transpose<<<grdMask, blk, 0, stream>>>(mask, maskT);

    // 2. q/k/v projections (WMMA): [65536,256] x [256,256]
    gemm_bf16_wmma<<<dim3(BN / 64), blk, 0, stream>>>(x_bf, wq_t, nullptr, q);
    gemm_bf16_wmma<<<dim3(BN / 64), blk, 0, stream>>>(x_bf, wk_t, nullptr, k);
    gemm_bf16_wmma<<<dim3(BN / 64), blk, 0, stream>>>(x_bf, wv_t, nullptr, v);

    // 3. mask-guided mechanism (two 1x1 convs as WMMA GEMMs + dw5 + sigmoid)
    gemm_bf16_wmma<<<dim3(NN / 64), blk, 0, stream>>>(maskT, w1_bf, mg_b1, m1);
    k_cast_bf16<<<grdMask, blk, 0, stream>>>(m1, m1_bf, NN * CC);
    gemm_bf16_wmma<<<dim3(NN / 64), blk, 0, stream>>>(m1_bf, w2_bf, mg_b2, m2);
    k_dw5_sigmoid_emb<<<grdMask, blk, 0, stream>>>(m1, m2, mg_dw, mg_db, emb);

    // 4. channel attention statistics
    k_colnorm_partial<<<dim3(BB * 64), blk, 0, stream>>>(q, k, accQ, accK);
    k_gram<<<dim3(BB * HEADS * 8), blk, 0, stream>>>(q, k, G);
    k_softmax<<<dim3(4), blk, 0, stream>>>(G, accQ, accK, rescale, attn);

    // 5. v scaling + per-head mix (q buffer -> v_s, k buffer -> out_t)
    float* vs   = q;
    float* outT = k;
    k_vscale<<<grdFull, blk, 0, stream>>>(v, emb, vs);
    k_headmix<<<grdFull, blk, 0, stream>>>(attn, vs, outT);

    // 6. output projection (WMMA) straight into d_out with bias
    __bf16* outT_bf = x_bf;                  // x_bf dead, reuse
    k_cast_bf16<<<grdFull, blk, 0, stream>>>(outT, outT_bf, ELN);
    gemm_bf16_wmma<<<dim3(BN / 64), blk, 0, stream>>>(outT_bf, wp_t, bproj, out);

    // 7. band embedding: dw3x3 -> GELU -> dw3x3, accumulate into d_out
    float* y1 = q;                           // v_s dead, reuse
    k_band1<<<grdFull, blk, 0, stream>>>(v, be_w1, y1);
    k_band2<<<grdFull, blk, 0, stream>>>(y1, be_w2, out);
}
